// DbrxMoE_26817775796593
// MI455X (gfx1250) — matrix operand order repaired and checked by reading, not compiled, not run on hardware
//
#include <hip/hip_runtime.h>
#include <math.h>

typedef __bf16 bf16;
typedef __attribute__((ext_vector_type(16))) __bf16 v16bf;
typedef __attribute__((ext_vector_type(8)))  __bf16 bf16x8;
typedef __attribute__((ext_vector_type(4)))  __bf16 bf16x4;
typedef __attribute__((ext_vector_type(8)))  float  v8f;
typedef __attribute__((ext_vector_type(4)))  float  f32x4;

constexpr int T = 1024;     // tokens (2*512)
constexpr int D = 768;      // model dim
constexpr int E = 16;       // experts
constexpr int I = 3072;     // ffn hidden
constexpr int TOPK = 4;

constexpr int TILE_M = 64;  // token rows per block
constexpr int TILE_N = 32;  // output cols per block
constexpr int KC     = 32;  // K chunk (matches wmma bf16 K)
constexpr int MT_MAX = T / TILE_M;   // 16 worst-case m-tiles per expert

// ---------------------------------------------------------------- utilities
__device__ __forceinline__ v16bf make_frag(bf16x8 lo, bf16x8 hi) {
    return __builtin_shufflevector(lo, hi, 0,1,2,3,4,5,6,7,8,9,10,11,12,13,14,15);
}
__device__ __forceinline__ bf16x4 cvt4(f32x4 v) {
    bf16x4 r;
    r[0] = (bf16)v[0]; r[1] = (bf16)v[1]; r[2] = (bf16)v[2]; r[3] = (bf16)v[3];
    return r;
}

// ------------------------------------------------------------- zero kernel
__global__ void k_zero(float* __restrict__ out, int* __restrict__ ctrl, int n) {
    int i = blockIdx.x * blockDim.x + threadIdx.x;
    if (i < n) out[i] = 0.f;
    if (blockIdx.x == 0 && threadIdx.x < 48) ctrl[threadIdx.x] = 0; // cnt, cnt2, bases
}

// ------------------------------------------------------------------ router
__global__ void k_router(const float* __restrict__ x, const float* __restrict__ rw,
                         int* __restrict__ cnt, int* __restrict__ topk_idx,
                         float* __restrict__ topk_w) {
    int t   = blockIdx.x;
    int tid = threadIdx.x;          // 256 threads: 16 per expert
    int e   = tid >> 4;
    int l16 = tid & 15;

    float p = 0.f;
    const float* xt = x + (size_t)t * D;
    const float* re = rw + (size_t)e * D;
    for (int d = l16; d < D; d += 16) p += xt[d] * re[d];

    __shared__ float part[E][17];
    __shared__ float logit[E];
    part[e][l16] = p;
    __syncthreads();
    if (tid < E) {
        float s = 0.f;
        #pragma unroll
        for (int j = 0; j < 16; ++j) s += part[tid][j];
        logit[tid] = s;
    }
    __syncthreads();
    if (tid == 0) {
        float prob[E];
        float mx = logit[0];
        for (int j = 1; j < E; ++j) mx = fmaxf(mx, logit[j]);
        float den = 0.f;
        for (int j = 0; j < E; ++j) { prob[j] = __expf(logit[j] - mx); den += prob[j]; }
        float inv = 1.f / den;
        for (int j = 0; j < E; ++j) prob[j] *= inv;

        int   sel[TOPK];
        float sw[TOPK];
        float wsum = 0.f;
        #pragma unroll
        for (int k = 0; k < TOPK; ++k) {
            int   bi = 0;
            float bv = -1.f;
            for (int j = 0; j < E; ++j)
                if (prob[j] > bv) { bv = prob[j]; bi = j; }
            sel[k] = bi; sw[k] = bv; wsum += bv;
            prob[bi] = -2.f;
        }
        float winv = 1.f / wsum;
        #pragma unroll
        for (int k = 0; k < TOPK; ++k) {
            topk_idx[t * TOPK + k] = sel[k];
            topk_w  [t * TOPK + k] = sw[k] * winv;
            atomicAdd(&cnt[sel[k]], 1);
        }
    }
}

// ----------------------------------------------------------- prefix scan
__global__ void k_scan(const int* __restrict__ cnt, int* __restrict__ bases) {
    if (threadIdx.x == 0) {
        int s = 0;
        for (int e = 0; e < E; ++e) { bases[e] = s; s += cnt[e]; }
    }
}

// -------------------------------------------------------------- scatter
__global__ void k_scatter(const int* __restrict__ topk_idx, const float* __restrict__ topk_w,
                          const int* __restrict__ bases, int* __restrict__ cnt2,
                          int* __restrict__ pair_tok, float* __restrict__ pair_w) {
    int idx = blockIdx.x * blockDim.x + threadIdx.x;   // 0..T*TOPK-1
    if (idx >= T * TOPK) return;
    int e    = topk_idx[idx];
    int slot = atomicAdd(&cnt2[e], 1);
    int pos  = bases[e] + slot;
    pair_tok[pos] = idx / TOPK;
    pair_w  [pos] = topk_w[idx];
}

// --------------------------------------------- stage 1: gate/up + SiLU-GLU
// grid.x = E*MT_MAX (expert, m-tile), grid.y = I/TILE_N, block = 256 (8 waves)
__global__ void k_gateup(const float* __restrict__ x, const float* __restrict__ w1,
                         const float* __restrict__ v1, const int* __restrict__ cnt,
                         const int* __restrict__ bases, const int* __restrict__ pair_tok,
                         bf16* __restrict__ act) {
    int e  = blockIdx.x / MT_MAX;
    int m0 = (blockIdx.x % MT_MAX) * TILE_M;
    int ne = cnt[e];
    if (m0 >= ne) return;
    int base = bases[e];
    int n0   = blockIdx.y * TILE_N;

    __shared__ alignas(16) bf16 As[2][TILE_M][KC];   // 8 KB
    __shared__ alignas(16) bf16 Bw[2][TILE_N][KC];   // 4 KB
    __shared__ alignas(16) bf16 Bv[2][TILE_N][KC];   // 4 KB

    int tid = threadIdx.x;

    // A loader: 4 threads per row; invalid rows clamp to a valid token
    // (their C rows are never stored, so real-but-wrong data is harmless)
    int am = tid >> 2;
    int ak = (tid & 3) * 4;
    int arow_idx = min(m0 + am, ne - 1);
    const float* xrow = x + (size_t)pair_tok[base + arow_idx] * D + ak;
    // B loader: 8 threads per row, one f32x4 each
    int bn = tid >> 3;
    int bk = (tid & 7) * 4;
    const float* w1row = w1 + (size_t)e * I * D + (size_t)(n0 + bn) * D + bk;
    const float* v1row = v1 + (size_t)e * I * D + (size_t)(n0 + bn) * D + bk;

    int wave = tid >> 5, lane = tid & 31;
    int lm = lane & 15, lh = lane >> 4;
    int msub = wave >> 1, nsub = wave & 1;

    v8f gacc = {0.f,0.f,0.f,0.f,0.f,0.f,0.f,0.f};
    v8f uacc = {0.f,0.f,0.f,0.f,0.f,0.f,0.f,0.f};

    // prologue: fill buffer 0
    {
        f32x4 a0 = *(const f32x4*)(xrow);
        f32x4 a1 = *(const f32x4*)(xrow + 16);
        f32x4 wv = *(const f32x4*)(w1row);
        f32x4 vv = *(const f32x4*)(v1row);
        *(bf16x4*)&As[0][am][ak]      = cvt4(a0);
        *(bf16x4*)&As[0][am][ak + 16] = cvt4(a1);
        *(bf16x4*)&Bw[0][bn][bk]      = cvt4(wv);
        *(bf16x4*)&Bv[0][bn][bk]      = cvt4(vv);
    }
    __syncthreads();

    constexpr int NIT = D / KC;   // 24
    int fm = msub * 16 + lm;
    int fn = nsub * 16 + lm;

    #pragma unroll 2
    for (int i = 0; i < NIT - 1; ++i) {
        int cur = i & 1, nxt = cur ^ 1;
        int k1 = (i + 1) * KC;
        // prefetch next chunk while WMMAs run on current
        f32x4 a0 = *(const f32x4*)(xrow + k1);
        f32x4 a1 = *(const f32x4*)(xrow + k1 + 16);
        f32x4 wv = *(const f32x4*)(w1row + k1);
        f32x4 vv = *(const f32x4*)(v1row + k1);

        v16bf afrag = make_frag(*(const bf16x8*)&As[cur][fm][16 * lh],
                                *(const bf16x8*)&As[cur][fm][16 * lh + 8]);
        v16bf wfrag = make_frag(*(const bf16x8*)&Bw[cur][fn][16 * lh],
                                *(const bf16x8*)&Bw[cur][fn][16 * lh + 8]);
        v16bf vfrag = make_frag(*(const bf16x8*)&Bv[cur][fn][16 * lh],
                                *(const bf16x8*)&Bv[cur][fn][16 * lh + 8]);
        gacc = __builtin_amdgcn_wmma_f32_16x16x32_bf16(false, afrag, false, wfrag,
                                                       (short)0, gacc, false, false);
        uacc = __builtin_amdgcn_wmma_f32_16x16x32_bf16(false, afrag, false, vfrag,
                                                       (short)0, uacc, false, false);

        *(bf16x4*)&As[nxt][am][ak]      = cvt4(a0);
        *(bf16x4*)&As[nxt][am][ak + 16] = cvt4(a1);
        *(bf16x4*)&Bw[nxt][bn][bk]      = cvt4(wv);
        *(bf16x4*)&Bv[nxt][bn][bk]      = cvt4(vv);
        __syncthreads();
    }
    {   // epilogue: last chunk
        int cur = (NIT - 1) & 1;
        v16bf afrag = make_frag(*(const bf16x8*)&As[cur][fm][16 * lh],
                                *(const bf16x8*)&As[cur][fm][16 * lh + 8]);
        v16bf wfrag = make_frag(*(const bf16x8*)&Bw[cur][fn][16 * lh],
                                *(const bf16x8*)&Bw[cur][fn][16 * lh + 8]);
        v16bf vfrag = make_frag(*(const bf16x8*)&Bv[cur][fn][16 * lh],
                                *(const bf16x8*)&Bv[cur][fn][16 * lh + 8]);
        gacc = __builtin_amdgcn_wmma_f32_16x16x32_bf16(false, afrag, false, wfrag,
                                                       (short)0, gacc, false, false);
        uacc = __builtin_amdgcn_wmma_f32_16x16x32_bf16(false, afrag, false, vfrag,
                                                       (short)0, uacc, false, false);
    }

    int n = n0 + nsub * 16 + lm;
    #pragma unroll
    for (int r = 0; r < 8; ++r) {
        int mrow = m0 + msub * 16 + r + 8 * lh;
        if (mrow < ne) {
            float g = gacc[r], u = uacc[r];
            float a = (g / (1.f + __expf(-g))) * u;   // silu(g) * u
            act[(size_t)(base + mrow) * I + n] = (bf16)a;
        }
    }
}

// ---------------------------------------- stage 2: down proj + weighted combine
// grid.x = E*MT_MAX, grid.y = D/TILE_N, block = 256 (8 waves)
__global__ void k_down(const bf16* __restrict__ act, const float* __restrict__ w2,
                       const int* __restrict__ cnt, const int* __restrict__ bases,
                       const int* __restrict__ pair_tok, const float* __restrict__ pair_w,
                       float* __restrict__ out) {
    int e  = blockIdx.x / MT_MAX;
    int m0 = (blockIdx.x % MT_MAX) * TILE_M;
    int ne = cnt[e];
    if (m0 >= ne) return;
    int base = bases[e];
    int n0   = blockIdx.y * TILE_N;

    __shared__ alignas(16) bf16 As[2][TILE_M][KC];   // 8 KB
    __shared__ alignas(16) bf16 Bs[2][TILE_N][KC];   // 4 KB
    __shared__ int   toks[TILE_M];
    __shared__ float tws [TILE_M];

    int tid = threadIdx.x;
    if (tid < TILE_M) {
        bool v = (m0 + tid < ne);
        toks[tid] = v ? pair_tok[base + m0 + tid] : 0;
        tws [tid] = v ? pair_w  [base + m0 + tid] : 0.f;
    }

    // A loader: 4 threads per row, 8 bf16 (16B) each; clamp invalid rows
    int am = tid >> 2;
    int ak = (tid & 3) * 8;
    int arow_idx = min(m0 + am, ne - 1);
    const bf16* arow = act + (size_t)(base + arow_idx) * I + ak;
    // B loader: 8 threads per row, one f32x4 each
    int bn = tid >> 3;
    int bk = (tid & 7) * 4;
    const float* w2row = w2 + (size_t)e * D * I + (size_t)(n0 + bn) * I + bk;

    int wave = tid >> 5, lane = tid & 31;
    int lm = lane & 15, lh = lane >> 4;
    int msub = wave >> 1, nsub = wave & 1;

    v8f acc = {0.f,0.f,0.f,0.f,0.f,0.f,0.f,0.f};

    {   // prologue
        bf16x8 av = *(const bf16x8*)(arow);
        f32x4  bv = *(const f32x4*)(w2row);
        *(bf16x8*)&As[0][am][ak] = av;
        *(bf16x4*)&Bs[0][bn][bk] = cvt4(bv);
    }
    __syncthreads();

    constexpr int NIT = I / KC;   // 96
    int fm = msub * 16 + lm;
    int fn = nsub * 16 + lm;

    #pragma unroll 2
    for (int i = 0; i < NIT - 1; ++i) {
        int cur = i & 1, nxt = cur ^ 1;
        int k1 = (i + 1) * KC;
        bf16x8 av = *(const bf16x8*)(arow + k1);
        f32x4  bv = *(const f32x4*)(w2row + k1);

        v16bf afrag = make_frag(*(const bf16x8*)&As[cur][fm][16 * lh],
                                *(const bf16x8*)&As[cur][fm][16 * lh + 8]);
        v16bf bfrag = make_frag(*(const bf16x8*)&Bs[cur][fn][16 * lh],
                                *(const bf16x8*)&Bs[cur][fn][16 * lh + 8]);
        acc = __builtin_amdgcn_wmma_f32_16x16x32_bf16(false, afrag, false, bfrag,
                                                      (short)0, acc, false, false);

        *(bf16x8*)&As[nxt][am][ak] = av;
        *(bf16x4*)&Bs[nxt][bn][bk] = cvt4(bv);
        __syncthreads();
    }
    {   // epilogue
        int cur = (NIT - 1) & 1;
        v16bf afrag = make_frag(*(const bf16x8*)&As[cur][fm][16 * lh],
                                *(const bf16x8*)&As[cur][fm][16 * lh + 8]);
        v16bf bfrag = make_frag(*(const bf16x8*)&Bs[cur][fn][16 * lh],
                                *(const bf16x8*)&Bs[cur][fn][16 * lh + 8]);
        acc = __builtin_amdgcn_wmma_f32_16x16x32_bf16(false, afrag, false, bfrag,
                                                      (short)0, acc, false, false);
    }

    int n = n0 + nsub * 16 + lm;
    #pragma unroll
    for (int r = 0; r < 8; ++r) {
        int ml = msub * 16 + r + 8 * lh;   // row within block tile
        int mrow = m0 + ml;
        if (mrow < ne) {
            float y = acc[r] * tws[ml];
            atomicAdd(&out[(size_t)toks[ml] * D + n], y);
        }
    }
}

// ---------------------------------------------------------------- launcher
extern "C" void kernel_launch(void* const* d_in, const int* in_sizes, int n_in,
                              void* d_out, int out_size, void* d_ws, size_t ws_size,
                              hipStream_t stream) {
    const float* x  = (const float*)d_in[0];   // [T, D]
    const float* rw = (const float*)d_in[1];   // [E, D]
    const float* w1 = (const float*)d_in[2];   // [E, I, D]
    const float* v1 = (const float*)d_in[3];   // [E, I, D]
    const float* w2 = (const float*)d_in[4];   // [E, D, I]
    float* out = (float*)d_out;

    char* w = (char*)d_ws;
    int*   cnt      = (int*)  (w + 0);
    int*   cnt2     = (int*)  (w + 64);
    int*   bases    = (int*)  (w + 128);
    int*   topk_idx = (int*)  (w + 256);
    float* topk_w   = (float*)(w + 256   + 16384);
    int*   pair_tok = (int*)  (w + 256   + 2 * 16384);
    float* pair_w   = (float*)(w + 256   + 3 * 16384);
    bf16*  act      = (bf16*) (w + 65792);                 // [T*TOPK, I] bf16, ~25 MB

    k_zero<<<(T * D + 255) / 256, 256, 0, stream>>>(out, cnt, T * D);
    k_router<<<T, 256, 0, stream>>>(x, rw, cnt, topk_idx, topk_w);
    k_scan<<<1, 32, 0, stream>>>(cnt, bases);
    k_scatter<<<(T * TOPK + 255) / 256, 256, 0, stream>>>(topk_idx, topk_w, bases, cnt2,
                                                          pair_tok, pair_w);
    k_gateup<<<dim3(E * MT_MAX, I / TILE_N), 256, 0, stream>>>(x, w1, v1, cnt, bases,
                                                               pair_tok, act);
    k_down<<<dim3(E * MT_MAX, D / TILE_N), 256, 0, stream>>>(act, w2, cnt, bases,
                                                             pair_tok, pair_w, out);
}
